// MyelinFlow_48112223649985
// MI455X (gfx1250) — matrix operand order repaired and checked by reading, not compiled, not run on hardware
//
#include <hip/hip_runtime.h>
#include <hip/hip_bf16.h>
#include <cstddef>
#include <cstdint>

typedef _Float16 h8   __attribute__((ext_vector_type(8)));
typedef _Float16 v16h __attribute__((ext_vector_type(16)));
typedef float    v8f  __attribute__((ext_vector_type(8)));

__device__ __forceinline__ float lrelu(float x) { return x > 0.f ? x : 0.1f * x; }

// ---------------------------------------------------------------------------
// Pack conv weights OIHW fp32 -> f16 [Opad][Kpad] (K-major, zero padded).
// ---------------------------------------------------------------------------
__global__ void pack_weights_k(const float* __restrict__ w, _Float16* __restrict__ o,
                               int O, int K, int Kpad, int Ntot) {
  int idx = blockIdx.x * blockDim.x + threadIdx.x;
  if (idx >= Ntot) return;
  int oo = idx / Kpad;
  int k  = idx - oo * Kpad;
  float v = (oo < O && k < K) ? w[(size_t)oo * K + k] : 0.f;
  o[idx] = (_Float16)v;
}

// ---------------------------------------------------------------------------
// Implicit-GEMM conv via WMMA f32<-f16.
//  M = OH*OW output pixels, N = out-channels (padded to 16), K = CI*KS*KS
//  (padded to 32).
//  Workgroup tile: M=64 x N=Opad(<=64).  8 waves = 4 N-tiles x 2 M-halves;
//  each wave owns TWO 16x16 accumulators (rows mhalf*32..+31), so per K-step
//  it loads ONE B fragment (two 16-byte pieces, 16 bytes apart, of packed f16 weights,
//  L2-resident) and issues TWO independent v_wmma_f32_16x16x32_f16 ops.
//  A (im2col) staged fp32->f16 in LDS via one ds_store_b128 per thread;
//  A fragments read back as ds_load_b128 pairs per the ISA 16-bit A layout.
// ---------------------------------------------------------------------------
template <int KS>
__global__ __launch_bounds__(256) void conv_wmma_k(
    const float* __restrict__ in, const _Float16* __restrict__ wt,
    const float* __restrict__ bias, float* __restrict__ out,
    int CI, int IH, int IW, int O, int Opad, int Kpad,
    int OH, int OW, int stride, int pad, int act) {
  constexpr int KS2 = KS * KS;
  const int K    = CI * KS2;
  const int Mtot = OH * OW;
  const int n      = blockIdx.y;
  const int mblock = blockIdx.x * 64;
  const int tid  = threadIdx.x;
  const int lane = tid & 31;
  const int wid  = tid >> 5;      // 0..7 (wave32)
  const int mhalf = wid & 1;      // which 32 of the 64 M rows
  const int ntile = wid >> 1;     // which 16 out-channels
  const bool active = (ntile * 16) < Opad;

  __shared__ __align__(16) _Float16 Alds[64][40];  // 64 M x 32 K, padded stride

  const float* inN = in + (size_t)n * CI * IH * IW;

  v8f acc0 = {}, acc1 = {};
  const int kt_num = Kpad >> 5;
  for (int kt = 0; kt < kt_num; ++kt) {
    const int k0 = kt << 5;
    __syncthreads();
    // ---- stage A tile: 64x32 f16, one h8 (ds_store_b128) per thread ----
    {
      const int m  = tid >> 2;          // 0..63
      const int kq = (tid & 3) * 8;     // 0,8,16,24
      const int s  = mblock + m;
      const int oy = s / OW;
      const int ox = s - oy * OW;
      h8 stg;
#pragma unroll
      for (int j = 0; j < 8; ++j) {
        const int ka = k0 + kq + j;
        float v = 0.f;
        if (s < Mtot && ka < K) {
          const int ci = ka / KS2;
          const int r  = ka - ci * KS2;
          const int kh = r / KS;
          const int kw = r - kh * KS;
          const int iy = oy * stride + kh - pad;
          const int ix = ox * stride + kw - pad;
          if (iy >= 0 && iy < IH && ix >= 0 && ix < IW)
            v = inN[((size_t)ci * IH + iy) * IW + ix];
        }
        stg[j] = (_Float16)v;
      }
      *(h8*)&Alds[m][kq] = stg;
    }
    __syncthreads();
    if (active) {
      // B fragment: lane (N = lane&15) reads K {8h..8h+7, 16+8h..16+8h+7}, reused twice
      const int o = ntile * 16 + (lane & 15);
      const _Float16* wp = wt + (size_t)o * Kpad + k0 + ((lane >> 4) * 8);
      h8 blo = *(const h8*)(wp);
      h8 bhi = *(const h8*)(wp + 16);
      v16h bfrag;
#pragma unroll
      for (int i = 0; i < 8; ++i) { bfrag[i] = blo[i]; bfrag[i + 8] = bhi[i]; }
      if (kt + 1 < kt_num) __builtin_prefetch(wp + 32, 0, 1);  // next K-tile

      // A fragments for the wave's two M-subtiles (ISA 16-bit A 16x32 layout)
      const int mrow0 = mhalf * 32 + (lane & 15);
      const int kset  = (lane >> 4) * 8;
      h8 a0lo = *(const h8*)&Alds[mrow0][kset];
      h8 a0hi = *(const h8*)&Alds[mrow0][kset + 16];
      h8 a1lo = *(const h8*)&Alds[mrow0 + 16][kset];
      h8 a1hi = *(const h8*)&Alds[mrow0 + 16][kset + 16];
      v16h af0, af1;
#pragma unroll
      for (int i = 0; i < 8; ++i) {
        af0[i] = a0lo[i]; af0[i + 8] = a0hi[i];
        af1[i] = a1lo[i]; af1[i + 8] = a1hi[i];
      }
      acc0 = __builtin_amdgcn_wmma_f32_16x16x32_f16(
          false, af0, false, bfrag, (short)0, acc0, false, false);
      acc1 = __builtin_amdgcn_wmma_f32_16x16x32_f16(
          false, af1, false, bfrag, (short)0, acc1, false, false);
    }
  }
  // ---- epilogue: bias + optional leaky-relu, C/D layout stores ----
  if (active) {
    const int o = ntile * 16 + (lane & 15);
    if (o < O) {
      const float bo = bias ? bias[o] : 0.f;
      float* outNO = out + ((size_t)n * O + o) * Mtot;
      const int mbase = mblock + mhalf * 32 + ((lane >> 4) * 8);
#pragma unroll
      for (int r = 0; r < 8; ++r) {
        const int m0 = mbase + r;
        if (m0 < Mtot) {
          float v = acc0[r] + bo;
          if (act) v = lrelu(v);
          outNO[m0] = v;
        }
        const int m1 = mbase + 16 + r;
        if (m1 < Mtot) {
          float v = acc1[r] + bo;
          if (act) v = lrelu(v);
          outNO[m1] = v;
        }
      }
    }
  }
}

// ---------------------------------------------------------------------------
// Faithful correlation (broadcast over 81 channels) fused with concat.
// out channels: [81 corr | C f1 | extraC extra]
// ---------------------------------------------------------------------------
__global__ void corr_concat_k(const float* __restrict__ f1, const float* __restrict__ f2,
                              const float* __restrict__ extra, int C, int HW,
                              int extraC, int Ctot, float invC, float* __restrict__ out) {
  const int n = blockIdx.y;
  const int s = blockIdx.x * blockDim.x + threadIdx.x;
  if (s >= HW) return;
  const float* p1 = f1 + (size_t)n * C * HW + s;
  const float* p2 = f2 + (size_t)n * C * HW + s;
  float sum = 0.f, asum = 0.f;
  for (int c = 0; c < C; ++c) {
    float a = p1[(size_t)c * HW];
    float b = p2[(size_t)c * HW];
    sum += a * b;
    asum += fabsf(a);
  }
  const float corr = (asum > 0.1f) ? sum * invC : 0.f;
  float* po = out + (size_t)n * Ctot * HW + s;
  for (int q = 0; q < 81; ++q) po[(size_t)q * HW] = corr;
  for (int c = 0; c < C; ++c) po[(size_t)(81 + c) * HW] = p1[(size_t)c * HW];
  if (extra) {
    const float* pe = extra + (size_t)n * extraC * HW + s;
    for (int e = 0; e < extraC; ++e) po[(size_t)(81 + C + e) * HW] = pe[(size_t)e * HW];
  }
}

// ---------------------------------------------------------------------------
// Border-clamped bilinear warp (grid_sample border, align_corners=True).
// ---------------------------------------------------------------------------
__global__ void warp_k(const float* __restrict__ x, const float* __restrict__ flow,
                       float* __restrict__ out, int C, int H, int W) {
  const int n = blockIdx.y;
  const int HW = H * W;
  const int s = blockIdx.x * blockDim.x + threadIdx.x;
  if (s >= HW) return;
  const int oy = s / W, ox = s - (s / W) * W;
  const float* fl = flow + (size_t)n * 2 * HW;
  const float u = fl[s], v = fl[(size_t)HW + s];
  const float px = fminf(fmaxf((float)ox + u, 0.f), (float)(W - 1));
  const float py = fminf(fmaxf((float)oy + v, 0.f), (float)(H - 1));
  const float x0 = floorf(px), y0 = floorf(py);
  const float fx = px - x0, fy = py - y0;
  const int x0i = (int)x0, y0i = (int)y0;
  const int x1i = min(x0i + 1, W - 1), y1i = min(y0i + 1, H - 1);
  const float w00 = (1.f - fx) * (1.f - fy), w01 = fx * (1.f - fy);
  const float w10 = (1.f - fx) * fy,         w11 = fx * fy;
  for (int c = 0; c < C; ++c) {
    const float* pc = x + ((size_t)n * C + c) * HW;
    float r = pc[y0i * W + x0i] * w00 + pc[y0i * W + x1i] * w01 +
              pc[y1i * W + x0i] * w10 + pc[y1i * W + x1i] * w11;
    out[((size_t)n * C + c) * HW + s] = r;
  }
}

// ---------------------------------------------------------------------------
// Bilinear x2 upsample (align_corners=False, edge clamp) * gain.
// ---------------------------------------------------------------------------
__global__ void upsample2_k(const float* __restrict__ in, float* __restrict__ out,
                            int C, int IH, int IW, float gain) {
  const int n = blockIdx.y;
  const int OH = IH * 2, OW = IW * 2;
  const int tot = C * OH * OW;
  const int idx = blockIdx.x * blockDim.x + threadIdx.x;
  if (idx >= tot) return;
  const int c = idx / (OH * OW);
  const int r = idx - c * OH * OW;
  const int oy = r / OW, ox = r - (r / OW) * OW;
  const float px = (ox + 0.5f) * 0.5f - 0.5f;
  const float py = (oy + 0.5f) * 0.5f - 0.5f;
  const float x0 = floorf(px), y0 = floorf(py);
  const float fx = px - x0, fy = py - y0;
  const int x0i = (int)x0, y0i = (int)y0;
  const int x0c = max(x0i, 0), x1c = min(x0i + 1, IW - 1);
  const int y0c = max(y0i, 0), y1c = min(y0i + 1, IH - 1);
  const float* p = in + ((size_t)n * C + c) * IH * IW;
  float v = p[y0c * IW + x0c] * (1.f - fx) * (1.f - fy) +
            p[y0c * IW + x1c] * fx * (1.f - fy) +
            p[y1c * IW + x0c] * (1.f - fx) * fy +
            p[y1c * IW + x1c] * fx * fy;
  out[((size_t)n * C + c) * OH * OW + r] = v * gain;
}

// ---------------------------------------------------------------------------
// Final: flow4 = up8 + res4 (128x128), bilinear x4 upsample, *4 -> 512x512.
// ---------------------------------------------------------------------------
__global__ void final_up4_k(const float* __restrict__ up8, const float* __restrict__ fl4,
                            float* __restrict__ out) {
  const int IH = 128, IW = 128, OH = 512, OW = 512;
  const int nc = blockIdx.y;  // n*2 + c
  const int s = blockIdx.x * blockDim.x + threadIdx.x;
  if (s >= OH * OW) return;
  const int oy = s / OW, ox = s - (s / OW) * OW;
  const float px = (ox + 0.5f) * 0.25f - 0.5f;
  const float py = (oy + 0.5f) * 0.25f - 0.5f;
  const float x0 = floorf(px), y0 = floorf(py);
  const float fx = px - x0, fy = py - y0;
  const int x0i = (int)x0, y0i = (int)y0;
  const int x0c = max(x0i, 0), x1c = min(x0i + 1, IW - 1);
  const int y0c = max(y0i, 0), y1c = min(y0i + 1, IH - 1);
  const float* a = up8 + (size_t)nc * IH * IW;
  const float* b = fl4 + (size_t)nc * IH * IW;
  auto S = [&](int yy, int xx) { int i = yy * IW + xx; return a[i] + b[i]; };
  float v = S(y0c, x0c) * (1.f - fx) * (1.f - fy) + S(y0c, x1c) * fx * (1.f - fy) +
            S(y1c, x0c) * (1.f - fx) * fy + S(y1c, x1c) * fx * fy;
  out[(size_t)nc * OH * OW + s] = v * 4.f;
}

// ---------------------------------------------------------------------------
extern "C" void kernel_launch(void* const* d_in, const int* in_sizes, int n_in,
                              void* d_out, int out_size, void* d_ws, size_t ws_size,
                              hipStream_t stream) {
  (void)in_sizes; (void)n_in; (void)out_size; (void)ws_size;
  const float* img1 = (const float*)d_in[0];
  const float* img2 = (const float*)d_in[1];
  char* ws = (char*)d_ws;
  constexpr int Bn = 8;

  size_t off = 0;
  auto alloc = [&](size_t bytes) -> size_t {
    size_t r = off;
    off = (off + bytes + 255) & ~(size_t)255;
    return r;
  };
  // ping-pong temps (each <= 56 MB so stage-to-stage traffic stays in L2)
  const size_t oT0  = alloc((size_t)Bn * 107 * 128 * 128 * 4);  // 56.1 MB
  const size_t oT1  = alloc((size_t)Bn * 16 * 256 * 256 * 4);   // 33.6 MB
  const size_t oF4a = alloc((size_t)Bn * 24 * 128 * 128 * 4);
  const size_t oF4b = alloc((size_t)Bn * 24 * 128 * 128 * 4);
  const size_t oF8a = alloc((size_t)Bn * 32 * 64 * 64 * 4);
  const size_t oF8b = alloc((size_t)Bn * 32 * 64 * 64 * 4);
  const size_t oFL8 = alloc((size_t)Bn * 2 * 64 * 64 * 4);
  const size_t oUP8 = alloc((size_t)Bn * 2 * 128 * 128 * 4);
  const size_t oFL4 = alloc((size_t)Bn * 2 * 128 * 128 * 4);

  struct L { int widx, bidx, O, CI, KS; int Opad, Kpad; size_t off; };
  L Ls[12] = {
      {2, 3, 16, 3, 7},    {4, 5, 16, 16, 3},   {6, 7, 24, 16, 3},
      {8, 9, 24, 24, 3},   {10, 11, 32, 24, 3}, {12, 13, 32, 32, 3},
      {14, 15, 64, 113, 3},{16, 17, 32, 64, 3}, {18, 19, 2, 32, 3},
      {20, 21, 64, 107, 3},{22, 23, 32, 64, 3}, {24, 25, 2, 32, 3}};
  for (int i = 0; i < 12; ++i) {
    L& l = Ls[i];
    l.Opad = (l.O + 15) & ~15;
    const int K = l.CI * l.KS * l.KS;
    l.Kpad = (K + 31) & ~31;
    l.off = alloc((size_t)l.Opad * l.Kpad * 2);
    const int Ntot = l.Opad * l.Kpad;
    pack_weights_k<<<(Ntot + 255) / 256, 256, 0, stream>>>(
        (const float*)d_in[l.widx], (_Float16*)(ws + l.off), l.O, K, l.Kpad, Ntot);
  }

  float* T0  = (float*)(ws + oT0);
  float* T1  = (float*)(ws + oT1);
  float* F4a = (float*)(ws + oF4a);
  float* F4b = (float*)(ws + oF4b);
  float* F8a = (float*)(ws + oF8a);
  float* F8b = (float*)(ws + oF8b);
  float* FL8 = (float*)(ws + oFL8);
  float* UP8 = (float*)(ws + oUP8);
  float* FL4 = (float*)(ws + oFL4);

  auto conv = [&](const float* inp, int CI, int IH, int IW, const L& l,
                  float* outp, int OH, int OW, int stride, int pad, int act) {
    dim3 grid((OH * OW + 63) / 64, Bn);
    const _Float16* wt = (const _Float16*)(ws + l.off);
    const float* bias = (const float*)d_in[l.bidx];
    if (l.KS == 3)
      conv_wmma_k<3><<<grid, 256, 0, stream>>>(inp, wt, bias, outp, CI, IH, IW,
                                               l.O, l.Opad, l.Kpad, OH, OW,
                                               stride, pad, act);
    else
      conv_wmma_k<7><<<grid, 256, 0, stream>>>(inp, wt, bias, outp, CI, IH, IW,
                                               l.O, l.Opad, l.Kpad, OH, OW,
                                               stride, pad, act);
  };

  // -------- encoder (both images) --------
  for (int im = 0; im < 2; ++im) {
    const float* img = im ? img2 : img1;
    float* F4 = im ? F4b : F4a;
    float* F8 = im ? F8b : F8a;
    conv(img, 3, 512, 512, Ls[0], T0, 256, 256, 2, 3, 1);
    conv(T0, 16, 256, 256, Ls[1], T1, 256, 256, 1, 1, 1);
    conv(T1, 16, 256, 256, Ls[2], T0, 128, 128, 2, 1, 1);
    conv(T0, 24, 128, 128, Ls[3], F4, 128, 128, 1, 1, 1);
    conv(F4, 24, 128, 128, Ls[4], T0, 64, 64, 2, 1, 1);
    conv(T0, 32, 64, 64, Ls[5], F8, 64, 64, 1, 1, 1);
  }

  // -------- 1/8 scale: corr + decoder --------
  corr_concat_k<<<dim3((4096 + 255) / 256, Bn), 256, 0, stream>>>(
      F8a, F8b, nullptr, 32, 4096, 0, 113, 1.f / 32.f, T0);
  conv(T0, 113, 64, 64, Ls[6], T1, 64, 64, 1, 1, 1);
  conv(T1, 64, 64, 64, Ls[7], T0, 64, 64, 1, 1, 1);
  conv(T0, 32, 64, 64, Ls[8], FL8, 64, 64, 1, 1, 0);
  upsample2_k<<<dim3((2 * 128 * 128 + 255) / 256, Bn), 256, 0, stream>>>(
      FL8, UP8, 2, 64, 64, 2.0f);

  // -------- 1/4 scale: warp + corr + decoder --------
  warp_k<<<dim3((16384 + 255) / 256, Bn), 256, 0, stream>>>(F4b, UP8, T1, 24, 128, 128);
  corr_concat_k<<<dim3((16384 + 255) / 256, Bn), 256, 0, stream>>>(
      F4a, T1, UP8, 24, 16384, 2, 107, 1.f / 24.f, T0);
  conv(T0, 107, 128, 128, Ls[9], T1, 128, 128, 1, 1, 1);
  conv(T1, 64, 128, 128, Ls[10], T0, 128, 128, 1, 1, 1);
  conv(T0, 32, 128, 128, Ls[11], FL4, 128, 128, 1, 1, 0);

  // -------- flow4 = up8 + res4, x4 bilinear, *4 --------
  final_up4_k<<<dim3((512 * 512 + 255) / 256, Bn * 2), 256, 0, stream>>>(
      UP8, FL4, (float*)d_out);
}
